// KERN_68015102099918
// MI455X (gfx1250) — matrix operand order repaired and checked by reading, not compiled, not run on hardware
//
#include <hip/hip_runtime.h>
#include <math.h>

typedef __attribute__((ext_vector_type(16))) _Float16 v16h;
typedef __attribute__((ext_vector_type(8)))  _Float16 v8h;
typedef __attribute__((ext_vector_type(4)))  _Float16 v4h;
typedef __attribute__((ext_vector_type(8)))  float    v8f;

// ---------------- problem constants ----------------
constexpr int N_OBJ  = 256;
constexpr int C_CLS  = 151;
constexpr int CPAD   = 160;              // padded class dim for einsum
constexpr int HDIM   = 512;
constexpr int OBJD   = 4096;
constexpr int ODIM   = 512;
constexpr int BIMG   = 8;
constexpr int TSTEPS = 3;
constexpr int CH     = 32;               // objects per chunk
constexpr int NCHUNK = N_OBJ / CH;       // 8
constexpr int MCH    = CH * C_CLS;       // 4832 rows per chunk (151 * 32)
constexpr int KCLS   = C_CLS * ODIM;     // 77312
constexpr int SK     = 16;               // split-K slices for classifier

// ---------------- wave / fragment helpers ----------------
struct LaneInfo { int lr, lh, koff, kb, rL; };
__device__ __forceinline__ LaneInfo lane_info() {
  int lane = threadIdx.x & 31;
  LaneInfo li;
  li.lr   = lane & 15;
  li.lh   = lane >> 4;
  li.koff = li.lh << 4;      // A-frag K offset (contiguous k = 16h + s)
  li.kb   = li.lh << 4;      // B-frag K offset (contiguous k = 16h + s, matches A)
  li.rL   = li.lh * 8;       // C/D row offset
  return li;
}

__device__ __forceinline__ v8f wmma32(v16h a, v16h b, v8f c) {
  return __builtin_amdgcn_wmma_f32_16x16x32_f16(false, a, false, b, (short)0, c,
                                                false, false);
}

__device__ __forceinline__ v16h join8(v8h lo, v8h hi) {
  return __builtin_shufflevector(lo, hi, 0,1,2,3,4,5,6,7,8,9,10,11,12,13,14,15);
}
// A fragment (16x32 MxK) from f16 row-major: p = &A[row] + k0 + koff(lane), 16 contiguous k.
__device__ __forceinline__ v16h frag_a_h(const _Float16* __restrict__ p) {
  return join8(*(const v8h*)p, *(const v8h*)(p + 8));
}

// Accumulate a K span for a 32(M) x 64(N) wave tile, all-f16 operands.
// a0/a1: A rows (lane-resolved) for row groups mBase+0..15 / +16..31.
// w0: weight base W[nBase+lr] + kb (B[k,n] = W[n,k], 16 contiguous f16 per lane).
__device__ __forceinline__ void mma_span_h(v8f acc[8],
                                           const _Float16* __restrict__ a0,
                                           const _Float16* __restrict__ a1,
                                           const _Float16* __restrict__ w0,
                                           size_t ldw, int K) {
  for (int k0 = 0; k0 < K; k0 += 32) {
    v16h A0 = frag_a_h(a0 + k0);
    v16h A1 = frag_a_h(a1 + k0);
#pragma unroll
    for (int j = 0; j < 4; ++j) {
      v16h b = *(const v16h*)(w0 + (size_t)(j * 16) * ldw + k0);
      acc[j]     = wmma32(A0, b, acc[j]);
      acc[4 + j] = wmma32(A1, b, acc[4 + j]);
    }
  }
}

__device__ __forceinline__ void acc_zero8(v8f acc[8]) {
  v8f z = {};
#pragma unroll
  for (int j = 0; j < 8; ++j) acc[j] = z;
}

// ---------------- prep kernels ----------------

// f32 -> f16 bulk cast (8 elements/thread)
__global__ __launch_bounds__(256) void k_cvt(const float* __restrict__ s,
                                             _Float16* __restrict__ d, int n8) {
  int i = blockIdx.x * 256 + threadIdx.x;
  if (i >= n8) return;
  const float4* s4 = (const float4*)s;
  float4 a = s4[2 * i], b = s4[2 * i + 1];
  v8h o;
  o[0]=(_Float16)a.x; o[1]=(_Float16)a.y; o[2]=(_Float16)a.z; o[3]=(_Float16)a.w;
  o[4]=(_Float16)b.x; o[5]=(_Float16)b.y; o[6]=(_Float16)b.z; o[7]=(_Float16)b.w;
  ((v8h*)d)[i] = o;
}

// matrix -> padded f16 MTL[c][k]=matrix[k][c], MTR[c][k]=matrix[c][k] (160x160)
__global__ __launch_bounds__(256) void k_prep_mtx(const float* __restrict__ M,
                                                  _Float16* __restrict__ L,
                                                  _Float16* __restrict__ R) {
  int i = blockIdx.x * 256 + threadIdx.x;
  if (i >= CPAD * CPAD) return;
  int c = i / CPAD, k = i % CPAD;
  float l = 0.f, r = 0.f;
  if (c < C_CLS && k < C_CLS) {
    l = M[(size_t)k * C_CLS + c];
    r = M[(size_t)c * C_CLS + k];
  }
  L[i] = (_Float16)l;
  R[i] = (_Float16)r;
}

// ---------------- compute kernels ----------------

// X = obj_fmaps @ Wproj^T + bproj  (writes f32 X and f16 XH)
__global__ __launch_bounds__(256) void k_proj(const _Float16* __restrict__ A,
                                              const _Float16* __restrict__ Wp,
                                              const float* __restrict__ bp,
                                              float* __restrict__ X,
                                              _Float16* __restrict__ XH) {
  int wid = blockIdx.x * 8 + (threadIdx.x >> 5);
  const int NT = HDIM / 64;
  if (wid >= (N_OBJ / 32) * NT) return;
  int mt = wid / NT, nt = wid % NT;
  LaneInfo li = lane_info();
  int mBase = mt * 32, nBase = nt * 64;
  v8f acc[8]; acc_zero8(acc);
  const _Float16* a0 = A + (size_t)(mBase + li.lr) * OBJD + li.koff;
  const _Float16* w0 = Wp + (size_t)(nBase + li.lr) * OBJD + li.kb;
  mma_span_h(acc, a0, a0 + (size_t)16 * OBJD, w0, OBJD, OBJD);
#pragma unroll
  for (int g = 0; g < 2; ++g)
#pragma unroll
    for (int j = 0; j < 4; ++j) {
      int col = nBase + j * 16 + li.lr;
      float bb = bp[col];
#pragma unroll
      for (int v = 0; v < 8; ++v) {
        int m = mBase + g * 16 + v + li.rL;
        float s = acc[g * 4 + j][v] + bb;
        X[(size_t)m * HDIM + col] = s;
        XH[(size_t)m * HDIM + col] = (_Float16)s;
      }
    }
}

// hidden[(n*C+c)] = X[n]  (f32 HID and f16 HIDH shadow)
__global__ __launch_bounds__(128) void k_init_hidden(const float* __restrict__ X,
                                                     float* __restrict__ HID,
                                                     _Float16* __restrict__ HIDH) {
  int row = blockIdx.x;                // < N*C
  int n = row / C_CLS;
  float4 v = ((const float4*)(X + (size_t)n * HDIM))[threadIdx.x];
  ((float4*)(HID + (size_t)row * HDIM))[threadIdx.x] = v;
  v4h h;
  h[0]=(_Float16)v.x; h[1]=(_Float16)v.y; h[2]=(_Float16)v.z; h[3]=(_Float16)v.w;
  ((v4h*)(HIDH + (size_t)row * HDIM))[threadIdx.x] = h;
}

// hs[b,c,h] = sum over n with im[n]==b of hidden[n,c,h]
__global__ __launch_bounds__(256) void k_segsum(const float* __restrict__ HID,
                                                const int* __restrict__ im,
                                                float* __restrict__ HS) {
  __shared__ int sim[N_OBJ];
  sim[threadIdx.x] = im[threadIdx.x];
  __syncthreads();
  int b = blockIdx.y;
  int e = blockIdx.x * 256 + threadIdx.x;          // < C*H
  const size_t CHH = (size_t)C_CLS * HDIM;
  float s = 0.f;
  for (int n = 0; n < N_OBJ; ++n)
    if (sim[n] == b) s += HID[(size_t)n * CHH + e];
  HS[(size_t)b * CHH + e] = s;
}

// DIFFT[nl][h][k(160)] = f16(hs[im,k,h] - hid[n,k,h]); zero-padded k>=151.
// LDS transpose: coalesced f32 reads (h-fastest) -> coalesced f16 writes (k-fastest)
__global__ __launch_bounds__(256) void k_diff(const float* __restrict__ HS,
                                              const float* __restrict__ HID,
                                              const int* __restrict__ im,
                                              _Float16* __restrict__ DT, int n0) {
  __shared__ _Float16 t[CPAD * 16];
  int nl = blockIdx.y;
  int h0 = blockIdx.x * 16;
  int n = n0 + nl;
  int b = im[n];
  const size_t CHH = (size_t)C_CLS * HDIM;
  const float* hsp = HS + (size_t)b * CHH;
  const float* hip = HID + (size_t)n * CHH;
  for (int i = threadIdx.x; i < CPAD * 16; i += 256) {
    int h = i & 15, k = i >> 4;
    float v = 0.f;
    if (k < C_CLS) {
      size_t o = (size_t)k * HDIM + h0 + h;
      v = hsp[o] - hip[o];
    }
    t[k * 16 + h] = (_Float16)v;
  }
  __syncthreads();
  _Float16* drow = DT + ((size_t)nl * HDIM + h0) * CPAD;
  for (int i = threadIdx.x; i < CPAD * 16; i += 256) {
    int k = i % CPAD, h = i / CPAD;
    drow[(size_t)h * CPAD + k] = t[k * 16 + h];
  }
}

// av(L/R) = MT(L/R)[160x160] @ diff^T -> AVH f16 [nl*C+c][2H]
__global__ __launch_bounds__(256) void k_einsum_h(const _Float16* __restrict__ MTL,
                                                  const _Float16* __restrict__ MTR,
                                                  const _Float16* __restrict__ DT,
                                                  _Float16* __restrict__ AVH) {
  int wid = blockIdx.x * 8 + (threadIdx.x >> 5);
  if (wid >= CH * 2 * 5 * 8) return;                 // 2560 waves
  int nl = wid / 80, rem = wid % 80;
  int sel = rem / 40; rem %= 40;
  int mt = rem / 8, nt = rem % 8;
  LaneInfo li = lane_info();
  int mBase = mt * 32, nBase = nt * 64;
  const _Float16* Am = sel ? MTR : MTL;
  const _Float16* a0 = Am + (size_t)(mBase + li.lr) * CPAD + li.koff;
  const _Float16* w0 = DT + ((size_t)nl * HDIM + nBase + li.lr) * CPAD + li.kb;
  v8f acc[8]; acc_zero8(acc);
  mma_span_h(acc, a0, a0 + (size_t)16 * CPAD, w0, CPAD, CPAD);
  int off = sel ? HDIM : 0;
#pragma unroll
  for (int g = 0; g < 2; ++g)
#pragma unroll
    for (int j = 0; j < 4; ++j) {
      int col = nBase + j * 16 + li.lr;
#pragma unroll
      for (int v = 0; v < 8; ++v) {
        int c = mBase + g * 16 + v + li.rL;
        if (c < C_CLS)
          AVH[((size_t)nl * C_CLS + c) * (2 * HDIM) + off + col] =
              (_Float16)acc[g * 4 + j][v];
      }
    }
}

// OUTB = sigmoid(AV @ Ww^T + hidden @ Wu^T + bw + bu)
__global__ __launch_bounds__(256) void k_gate(const _Float16* __restrict__ AVH,
                                              const _Float16* __restrict__ HIDH,
                                              const _Float16* __restrict__ Ww,
                                              const float* __restrict__ bw,
                                              const _Float16* __restrict__ Wu,
                                              const float* __restrict__ bu,
                                              float* __restrict__ OUTB, int n0) {
  int wid = blockIdx.x * 8 + (threadIdx.x >> 5);
  const int NT = HDIM / 64;
  if (wid >= (MCH / 32) * NT) return;
  int mt = wid / NT, nt = wid % NT;
  LaneInfo li = lane_info();
  int mBase = mt * 32, nBase = nt * 64;
  v8f acc[8]; acc_zero8(acc);
  const _Float16* a0 = AVH + (size_t)(mBase + li.lr) * (2 * HDIM) + li.koff;
  const _Float16* w1 = Ww + (size_t)(nBase + li.lr) * (2 * HDIM) + li.kb;
  mma_span_h(acc, a0, a0 + (size_t)16 * (2 * HDIM), w1, 2 * HDIM, 2 * HDIM);
  const _Float16* a2 =
      HIDH + ((size_t)n0 * C_CLS + mBase + li.lr) * HDIM + li.koff;
  const _Float16* w2 = Wu + (size_t)(nBase + li.lr) * HDIM + li.kb;
  mma_span_h(acc, a2, a2 + (size_t)16 * HDIM, w2, HDIM, HDIM);
#pragma unroll
  for (int g = 0; g < 2; ++g)
#pragma unroll
    for (int j = 0; j < 4; ++j) {
      int col = nBase + j * 16 + li.lr;
      float bb = bw[col] + bu[col];
#pragma unroll
      for (int v = 0; v < 8; ++v) {
        int m = mBase + g * 16 + v + li.rL;
        float s = acc[g * 4 + j][v] + bb;
        OUTB[(size_t)m * HDIM + col] = 1.f / (1.f + expf(-s));
      }
    }
}

// RVHH = f16(rv * hidden_chunk)
__global__ __launch_bounds__(256) void k_rh(const float* __restrict__ RV,
                                            const float* __restrict__ HID,
                                            _Float16* __restrict__ RVHH, int n0) {
  size_t e = (size_t)blockIdx.x * 256 + threadIdx.x;   // < MCH*H/4
  const float4* r4 = (const float4*)RV;
  const float4* h4 = (const float4*)(HID + (size_t)n0 * C_CLS * HDIM);
  float4 r = r4[e], h = h4[e];
  v4h o;
  o[0]=(_Float16)(r.x*h.x); o[1]=(_Float16)(r.y*h.y);
  o[2]=(_Float16)(r.z*h.z); o[3]=(_Float16)(r.w*h.w);
  ((v4h*)RVHH)[e] = o;
}

// hv = tanh(AV@W5w^T + RVH@W5u^T + b); hidden = (1-z)*hidden + z*hv (in place,
// both f32 HID and f16 HIDH shadow)
__global__ __launch_bounds__(256) void k_hv_upd(const _Float16* __restrict__ AVH,
                                                const _Float16* __restrict__ RVHH,
                                                const _Float16* __restrict__ Ww,
                                                const float* __restrict__ bw,
                                                const _Float16* __restrict__ Wu,
                                                const float* __restrict__ bu,
                                                const float* __restrict__ ZV,
                                                float* __restrict__ HID,
                                                _Float16* __restrict__ HIDH, int n0) {
  int wid = blockIdx.x * 8 + (threadIdx.x >> 5);
  const int NT = HDIM / 64;
  if (wid >= (MCH / 32) * NT) return;
  int mt = wid / NT, nt = wid % NT;
  LaneInfo li = lane_info();
  int mBase = mt * 32, nBase = nt * 64;
  v8f acc[8]; acc_zero8(acc);
  const _Float16* a0 = AVH + (size_t)(mBase + li.lr) * (2 * HDIM) + li.koff;
  const _Float16* w1 = Ww + (size_t)(nBase + li.lr) * (2 * HDIM) + li.kb;
  mma_span_h(acc, a0, a0 + (size_t)16 * (2 * HDIM), w1, 2 * HDIM, 2 * HDIM);
  const _Float16* a2 = RVHH + (size_t)(mBase + li.lr) * HDIM + li.koff;
  const _Float16* w2 = Wu + (size_t)(nBase + li.lr) * HDIM + li.kb;
  mma_span_h(acc, a2, a2 + (size_t)16 * HDIM, w2, HDIM, HDIM);
#pragma unroll
  for (int g = 0; g < 2; ++g)
#pragma unroll
    for (int j = 0; j < 4; ++j) {
      int col = nBase + j * 16 + li.lr;
      float bb = bw[col] + bu[col];
#pragma unroll
      for (int v = 0; v < 8; ++v) {
        int m = mBase + g * 16 + v + li.rL;
        float hv = tanhf(acc[g * 4 + j][v] + bb);
        size_t lidx = (size_t)m * HDIM + col;
        size_t gidx = ((size_t)n0 * C_CLS + m) * HDIM + col;
        float z = ZV[lidx];
        float nh = (1.f - z) * HID[gidx] + z * hv;
        HID[gidx] = nh;
        HIDH[gidx] = (_Float16)nh;
      }
    }
}

// OUTCH = f16(relu([hidden ; inp] @ Wout^T + bout)), chunk rows
__global__ __launch_bounds__(256) void k_out(const _Float16* __restrict__ HIDH,
                                             const _Float16* __restrict__ XH,
                                             const _Float16* __restrict__ Wo,
                                             const float* __restrict__ bo,
                                             _Float16* __restrict__ OUTCH, int n0) {
  int wid = blockIdx.x * 8 + (threadIdx.x >> 5);
  const int NT = ODIM / 64;
  if (wid >= (MCH / 32) * NT) return;
  int mt = wid / NT, nt = wid % NT;
  LaneInfo li = lane_info();
  int mBase = mt * 32, nBase = nt * 64;
  v8f acc[8]; acc_zero8(acc);
  int r0 = mBase + li.lr, r1 = r0 + 16;
  const _Float16* a0 = HIDH + ((size_t)n0 * C_CLS + r0) * HDIM + li.koff;
  const _Float16* w1 = Wo + (size_t)(nBase + li.lr) * (2 * HDIM) + li.kb;
  mma_span_h(acc, a0, a0 + (size_t)16 * HDIM, w1, 2 * HDIM, HDIM);
  const _Float16* x0 = XH + (size_t)(n0 + r0 / C_CLS) * HDIM + li.koff;   // inp
  const _Float16* x1 = XH + (size_t)(n0 + r1 / C_CLS) * HDIM + li.koff;
  mma_span_h(acc, x0, x1, w1 + HDIM, 2 * HDIM, HDIM);
#pragma unroll
  for (int g = 0; g < 2; ++g)
#pragma unroll
    for (int j = 0; j < 4; ++j) {
      int col = nBase + j * 16 + li.lr;
      float bb = bo[col];
#pragma unroll
      for (int v = 0; v < 8; ++v) {
        int m = mBase + g * 16 + v + li.rL;
        float s = acc[g * 4 + j][v] + bb;
        OUTCH[(size_t)m * ODIM + col] = (_Float16)(s > 0.f ? s : 0.f);
      }
    }
}

// split-K classifier partials over K=77312; 32 objects, 151 cols
__global__ __launch_bounds__(256) void k_cls(const _Float16* __restrict__ OUTCH,
                                             const _Float16* __restrict__ Wc,
                                             float* __restrict__ PART) {
  int wid = blockIdx.x * 8 + (threadIdx.x >> 5);
  if (wid >= 3 * SK) return;
  int ks = wid / 3, nt = wid % 3;
  LaneInfo li = lane_info();
  int nBase = nt * 64;
  const _Float16* a0 = OUTCH + (size_t)li.lr * KCLS + li.koff;
  const _Float16* a1 = a0 + (size_t)16 * KCLS;
  int colB = nBase + li.lr;
  v8f acc[8]; acc_zero8(acc);
  int kstart = ks * (KCLS / SK);
  for (int k0 = kstart; k0 < kstart + KCLS / SK; k0 += 32) {
    v16h A0 = frag_a_h(a0 + k0);
    v16h A1 = frag_a_h(a1 + k0);
#pragma unroll
    for (int j = 0; j < 4; ++j) {
      int col = colB + j * 16;
      v16h b = {};
      if (col < C_CLS) b = *(const v16h*)(Wc + (size_t)col * KCLS + li.kb + k0);
      acc[j]     = wmma32(A0, b, acc[j]);
      acc[4 + j] = wmma32(A1, b, acc[4 + j]);
    }
  }
#pragma unroll
  for (int g = 0; g < 2; ++g)
#pragma unroll
    for (int j = 0; j < 4; ++j) {
      int col = nBase + j * 16 + li.lr;
      if (col < C_CLS) {
#pragma unroll
        for (int v = 0; v < 8; ++v) {
          int m = g * 16 + v + li.rL;
          PART[((size_t)ks * CH + m) * C_CLS + col] = acc[g * 4 + j][v];
        }
      }
    }
}

__global__ __launch_bounds__(256) void k_cls_reduce(const float* __restrict__ PART,
                                                    const float* __restrict__ bc,
                                                    float* __restrict__ dout, int n0) {
  int e = blockIdx.x * 256 + threadIdx.x;
  if (e >= CH * C_CLS) return;
  int m = e / C_CLS, c = e % C_CLS;
  float s = bc[c];
#pragma unroll
  for (int ks = 0; ks < SK; ++ks) s += PART[((size_t)ks * CH + m) * C_CLS + c];
  dout[(size_t)(n0 + m) * C_CLS + c] = s;
}

// ---------------- host launcher ----------------
extern "C" void kernel_launch(void* const* d_in, const int* in_sizes, int n_in,
                              void* d_out, int out_size, void* d_ws, size_t ws_size,
                              hipStream_t stream) {
  (void)in_sizes; (void)n_in; (void)out_size;
  const float* obj = (const float*)d_in[0];
  const int*   im  = (const int*)d_in[1];
  const float* mtx = (const float*)d_in[2];
  const float* Wp  = (const float*)d_in[3];
  const float* bp  = (const float*)d_in[4];
  const float* W3w = (const float*)d_in[5];
  const float* b3w = (const float*)d_in[6];
  const float* W3u = (const float*)d_in[7];
  const float* b3u = (const float*)d_in[8];
  const float* W4w = (const float*)d_in[9];
  const float* b4w = (const float*)d_in[10];
  const float* W4u = (const float*)d_in[11];
  const float* b4u = (const float*)d_in[12];
  const float* W5w = (const float*)d_in[13];
  const float* b5w = (const float*)d_in[14];
  const float* W5u = (const float*)d_in[15];
  const float* b5u = (const float*)d_in[16];
  const float* Wo  = (const float*)d_in[17];
  const float* bo  = (const float*)d_in[18];
  const float* Wc  = (const float*)d_in[19];
  const float* bc  = (const float*)d_in[20];
  float* dout = (float*)d_out;

  // ---- f32 workspace region ----
  float* base = (float*)d_ws;
  float* X    = base;                                   // [256,512]
  float* HS   = X   + (size_t)N_OBJ * HDIM;             // [8,151,512]
  float* HID  = HS  + (size_t)BIMG * C_CLS * HDIM;      // [38656,512]
  float* ZV   = HID + (size_t)N_OBJ * C_CLS * HDIM;     // [4832,512]
  float* RV   = ZV  + (size_t)MCH * HDIM;               // [4832,512]
  float* PART = RV  + (size_t)MCH * HDIM;               // [16,32,151]
  // ---- f16 workspace region (32B aligned by construction) ----
  _Float16* OBJH  = (_Float16*)(PART + (size_t)SK * CH * C_CLS);
  _Float16* WpH   = OBJH + (size_t)N_OBJ * OBJD;
  _Float16* W3wH  = WpH  + (size_t)HDIM * OBJD;
  _Float16* W3uH  = W3wH + (size_t)HDIM * 2 * HDIM;
  _Float16* W4wH  = W3uH + (size_t)HDIM * HDIM;
  _Float16* W4uH  = W4wH + (size_t)HDIM * 2 * HDIM;
  _Float16* W5wH  = W4uH + (size_t)HDIM * HDIM;
  _Float16* W5uH  = W5wH + (size_t)HDIM * 2 * HDIM;
  _Float16* WoH   = W5uH + (size_t)HDIM * HDIM;
  _Float16* WcH   = WoH  + (size_t)ODIM * 2 * HDIM;
  _Float16* MTL   = WcH  + (size_t)C_CLS * KCLS;
  _Float16* MTR   = MTL  + (size_t)CPAD * CPAD;
  _Float16* XH    = MTR  + (size_t)CPAD * CPAD;
  _Float16* HIDH  = XH   + (size_t)N_OBJ * HDIM;
  _Float16* AVH   = HIDH + (size_t)N_OBJ * C_CLS * HDIM;
  _Float16* RVHH  = AVH  + (size_t)MCH * 2 * HDIM;
  _Float16* OUTCH = RVHH + (size_t)MCH * HDIM;
  _Float16* DIFFT = OUTCH + (size_t)MCH * ODIM;
  _Float16* wend  = DIFFT + (size_t)CH * HDIM * CPAD;
  if (ws_size < (size_t)((char*)wend - (char*)base)) return;  // ~205 MB

  // ---- one-time f16 staging of all WMMA operand tensors ----
  auto cvt = [&](const float* s, _Float16* d, size_t n) {
    int n8 = (int)(n / 8);
    k_cvt<<<(n8 + 255) / 256, 256, 0, stream>>>(s, d, n8);
  };
  cvt(obj, OBJH, (size_t)N_OBJ * OBJD);
  cvt(Wp,  WpH,  (size_t)HDIM * OBJD);
  cvt(W3w, W3wH, (size_t)HDIM * 2 * HDIM);
  cvt(W3u, W3uH, (size_t)HDIM * HDIM);
  cvt(W4w, W4wH, (size_t)HDIM * 2 * HDIM);
  cvt(W4u, W4uH, (size_t)HDIM * HDIM);
  cvt(W5w, W5wH, (size_t)HDIM * 2 * HDIM);
  cvt(W5u, W5uH, (size_t)HDIM * HDIM);
  cvt(Wo,  WoH,  (size_t)ODIM * 2 * HDIM);
  cvt(Wc,  WcH,  (size_t)C_CLS * KCLS);
  k_prep_mtx<<<(CPAD * CPAD + 255) / 256, 256, 0, stream>>>(mtx, MTL, MTR);

  k_proj<<<(N_OBJ / 32) * (HDIM / 64) / 8, 256, 0, stream>>>(OBJH, WpH, bp, X, XH);
  k_init_hidden<<<N_OBJ * C_CLS, 128, 0, stream>>>(X, HID, HIDH);

  for (int t = 0; t < TSTEPS; ++t) {
    k_segsum<<<dim3((C_CLS * HDIM) / 256, BIMG), 256, 0, stream>>>(HID, im, HS);
    for (int ch = 0; ch < NCHUNK; ++ch) {
      int n0 = ch * CH;
      k_diff<<<dim3(HDIM / 16, CH), 256, 0, stream>>>(HS, HID, im, DIFFT, n0);
      k_einsum_h<<<(CH * 2 * 5 * 8) / 8, 256, 0, stream>>>(MTL, MTR, DIFFT, AVH);
      k_gate<<<MCH / 32, 256, 0, stream>>>(AVH, HIDH, W3wH, b3w, W3uH, b3u, ZV, n0);
      k_gate<<<MCH / 32, 256, 0, stream>>>(AVH, HIDH, W4wH, b4w, W4uH, b4u, RV, n0);
      k_rh<<<(MCH * HDIM / 4) / 256, 256, 0, stream>>>(RV, HID, RVHH, n0);
      k_hv_upd<<<MCH / 32, 256, 0, stream>>>(AVH, RVHH, W5wH, b5w, W5uH, b5u, ZV,
                                             HID, HIDH, n0);
    }
  }
  for (int ch = 0; ch < NCHUNK; ++ch) {
    int n0 = ch * CH;
    k_out<<<MCH / 32, 256, 0, stream>>>(HIDH, XH, WoH, bo, OUTCH, n0);
    k_cls<<<(3 * SK) / 8, 256, 0, stream>>>(OUTCH, WcH, PART);
    k_cls_reduce<<<(CH * C_CLS + 255) / 256, 256, 0, stream>>>(PART, bc, dout, n0);
  }
}
